// SelfCorrelationStack_19782619365605
// MI455X (gfx1250) — matrix operand order repaired and checked by reading, not compiled, not run on hardware
//
#include <hip/hip_runtime.h>

// ---------------------------------------------------------------------------
// Grouped per-stack MLP on gfx1250 (MI455X): h = relu(x@W1+b1); w = h@W2+b2;
// out = lam*x*w + x.   S=8, rows=N*I=65536, D=H=64.
// Memory-bound (AI ~32 FLOP/B): x read once, out written once; weights in
// bf16 WMMA fragments resident in VGPRs; LDS only for layout transposes.
// ---------------------------------------------------------------------------

typedef __attribute__((ext_vector_type(16))) __bf16 v16bf;
typedef __attribute__((ext_vector_type(8)))  __bf16 v8bf;
typedef __attribute__((ext_vector_type(8)))  float  v8f;
typedef __attribute__((ext_vector_type(4)))  float  v4f;

#define S_DIM            8
#define ROWS_PER_STACK   65536     // N * I
#define D_DIM            64
#define BLOCKS_PER_STACK 64
#define ROWS_PER_BLOCK   (ROWS_PER_STACK / BLOCKS_PER_STACK)  // 1024
#define WAVES            8
#define ROWS_PER_ITER    (WAVES * 16)                          // 128
#define N_ITERS          (ROWS_PER_BLOCK / ROWS_PER_ITER)      // 8

static __device__ __forceinline__ v8f wmma_bf16(v16bf a, v16bf b, v8f c) {
  // 8 args: (neg_a, A, neg_b, B, c_mod, C, reuse_a, reuse_b)
  return __builtin_amdgcn_wmma_f32_16x16x32_bf16(false, a, false, b,
                                                 (short)0, c, false, false);
}

#define SHUF16(lo, hi) \
  __builtin_shufflevector(lo, hi, 0,1,2,3,4,5,6,7,8,9,10,11,12,13,14,15)

// Gather a 64x64 fp32 weight matrix (K-major, row stride 64) into bf16 WMMA
// B-fragments. Fragment f = kstep*4 + ntile. B layout (32x16 bf16, wave32):
// lane n%16 holds 16 K values for its column; lanes 0-15 cover
// K {0-7,16-23} (+kbase), lanes 16-31 cover K {8-15,24-31} (+kbase).
static __device__ __forceinline__ void stage_frags(__bf16 (*frag)[32][16],
                                                   const float* __restrict__ W,
                                                   int tid) {
  const int f     = tid >> 5;          // 8 fragments
  const int ln    = tid & 31;          // 32 lanes' worth of data
  const int kstep = f >> 2;
  const int nt    = f & 3;
  const int n     = nt * 16 + (ln & 15);
  const int kb    = kstep * 32 + ((ln & 16) ? 8 : 0);
#pragma unroll
  for (int j = 0; j < 16; ++j)
    frag[f][ln][j] = (__bf16)W[(kb + j + (j & 8)) * 64 + n];
}

__global__ __launch_bounds__(256) void grouped_mlp_kernel(
    const float* __restrict__ x,  const float* __restrict__ W1,
    const float* __restrict__ b1, const float* __restrict__ W2,
    const float* __restrict__ b2, const float* __restrict__ lam,
    float* __restrict__ out)
{
  __shared__ __bf16 fragStage[8][32][16];   //  8 KB, reused for W1 then W2
  __shared__ __bf16 hstage[WAVES][16 * 64]; // 16 KB, per-wave h transpose
  __shared__ float  wstage[WAVES][16 * 64]; // 32 KB, per-wave w transpose

  const int tid   = threadIdx.x;
  const int wave  = tid >> 5;
  const int lane  = tid & 31;
  const int s     = blockIdx.x / BLOCKS_PER_STACK;
  const int btile = blockIdx.x % BLOCKS_PER_STACK;

  // ---- stage weights to bf16 fragments, hoist into VGPRs -----------------
  stage_frags(fragStage, W1 + s * 64 * 64, tid);
  __syncthreads();
  v16bf B1f[8];
#pragma unroll
  for (int f = 0; f < 8; ++f) B1f[f] = *(const v16bf*)&fragStage[f][lane][0];
  __syncthreads();
  stage_frags(fragStage, W2 + s * 64 * 64, tid);
  __syncthreads();
  v16bf B2f[8];
#pragma unroll
  for (int f = 0; f < 8; ++f) B2f[f] = *(const v16bf*)&fragStage[f][lane][0];

  float b1v[4], b2v[4];
#pragma unroll
  for (int nt = 0; nt < 4; ++nt) {
    b1v[nt] = b1[s * 64 + nt * 16 + (lane & 15)];
    b2v[nt] = b2[s * 64 + nt * 16 + (lane & 15)];
  }
  const float lamv = lam[s];

  // A-layout constants (16x32 bf16): lanes 0-15 take K {0-7,16-23} (+kstep*32),
  // lanes 16-31 take K {8-15,24-31}; row = lane & 15.
  const int  r     = lane & 15;
  const bool hiL   = lane >= 16;
  const int  koff  = hiL ? 8 : 0;
  const int  mbase = hiL ? 8 : 0;   // C-layout: VGPR j -> M = j + mbase

  __bf16* hbuf = &hstage[wave][0];
  float*  wbuf = &wstage[wave][0];
  const long stackBase = (long)s * ROWS_PER_STACK;

  for (int it = 0; it < N_ITERS; ++it) {
    const long row0 = (long)btile * ROWS_PER_BLOCK + it * ROWS_PER_ITER + wave * 16;
    const float* xrow = x   + (stackBase + row0 + r) * D_DIM;
    float*       orow = out + (stackBase + row0 + r) * D_DIM;

    if (it + 1 < N_ITERS)                       // pull next tile toward L2/L0
      __builtin_prefetch(xrow + ROWS_PER_ITER * D_DIM, 0, 0);

    // ---- load x rows once (fp32 kept in VGPRs) + build bf16 A-fragments --
    v8f  xa[2], xb[2];   // xa = K[kb..kb+7], xb = K[kb+16..kb+23] as fp32
    v16bf A[2];
#pragma unroll
    for (int ks = 0; ks < 2; ++ks) {
      const int kb = ks * 32 + koff;
      v4f p0 = *(const v4f*)(xrow + kb);
      v4f p1 = *(const v4f*)(xrow + kb + 4);
      v4f p2 = *(const v4f*)(xrow + kb + 16);
      v4f p3 = *(const v4f*)(xrow + kb + 20);
      v8f lo = __builtin_shufflevector(p0, p1, 0,1,2,3,4,5,6,7);
      v8f hi = __builtin_shufflevector(p2, p3, 0,1,2,3,4,5,6,7);
      xa[ks] = lo;  xb[ks] = hi;
      v8bf cl = __builtin_convertvector(lo, v8bf);
      v8bf ch = __builtin_convertvector(hi, v8bf);
      A[ks] = SHUF16(cl, ch);
    }

    // ---- GEMM1: (16x64) x (64x64) -> h, 8 WMMAs ---------------------------
    v8f hacc[4];
#pragma unroll
    for (int nt = 0; nt < 4; ++nt) {
      v8f c = {};
      c = wmma_bf16(A[0], B1f[nt],     c);
      c = wmma_bf16(A[1], B1f[4 + nt], c);
      hacc[nt] = c;
    }

    // ---- bias + relu; C-layout -> LDS (bf16) ------------------------------
#pragma unroll
    for (int nt = 0; nt < 4; ++nt) {
      const int col = nt * 16 + r;
#pragma unroll
      for (int j = 0; j < 8; ++j) {
        float v = hacc[nt][j] + b1v[nt];
        v = v > 0.0f ? v : 0.0f;
        hbuf[(mbase + j) * 64 + col] = (__bf16)v;
      }
    }

    // ---- re-read h as A-fragments (DS in-order within wave; no barrier) ---
    v16bf HA[2];
#pragma unroll
    for (int ks = 0; ks < 2; ++ks) {
      const int kb = ks * 32 + koff;
      v8bf lo = *(const v8bf*)(hbuf + r * 64 + kb);
      v8bf hi = *(const v8bf*)(hbuf + r * 64 + kb + 16);
      HA[ks] = SHUF16(lo, hi);
    }

    // ---- GEMM2: h x W2 -> w, 8 WMMAs --------------------------------------
    v8f wacc[4];
#pragma unroll
    for (int nt = 0; nt < 4; ++nt) {
      v8f c = {};
      c = wmma_bf16(HA[0], B2f[nt],     c);
      c = wmma_bf16(HA[1], B2f[4 + nt], c);
      wacc[nt] = c;
    }

    // ---- + b2; C-layout -> LDS (fp32) -------------------------------------
#pragma unroll
    for (int nt = 0; nt < 4; ++nt) {
      const int col = nt * 16 + r;
#pragma unroll
      for (int j = 0; j < 8; ++j)
        wbuf[(mbase + j) * 64 + col] = wacc[nt][j] + b2v[nt];
    }

    // ---- fused epilogue in row-layout: out = x*(lam*w) + x ----------------
#pragma unroll
    for (int ks = 0; ks < 2; ++ks) {
      const int kb = ks * 32 + koff;
      v4f w0 = *(const v4f*)(wbuf + r * 64 + kb);
      v4f w1 = *(const v4f*)(wbuf + r * 64 + kb + 4);
      v4f w2 = *(const v4f*)(wbuf + r * 64 + kb + 16);
      v4f w3 = *(const v4f*)(wbuf + r * 64 + kb + 20);
      v4f o0, o1, o2, o3;
#pragma unroll
      for (int j = 0; j < 4; ++j) {
        o0[j] = fmaf(xa[ks][j],     lamv * w0[j], xa[ks][j]);
        o1[j] = fmaf(xa[ks][j + 4], lamv * w1[j], xa[ks][j + 4]);
        o2[j] = fmaf(xb[ks][j],     lamv * w2[j], xb[ks][j]);
        o3[j] = fmaf(xb[ks][j + 4], lamv * w3[j], xb[ks][j + 4]);
      }
      *(v4f*)(orow + kb)      = o0;
      *(v4f*)(orow + kb + 4)  = o1;
      *(v4f*)(orow + kb + 16) = o2;
      *(v4f*)(orow + kb + 20) = o3;
    }
  }
}

extern "C" void kernel_launch(void* const* d_in, const int* in_sizes, int n_in,
                              void* d_out, int out_size, void* d_ws, size_t ws_size,
                              hipStream_t stream) {
  const float* x   = (const float*)d_in[0];
  const float* W1  = (const float*)d_in[1];
  const float* b1  = (const float*)d_in[2];
  const float* W2  = (const float*)d_in[3];
  const float* b2  = (const float*)d_in[4];
  const float* lam = (const float*)d_in[5];
  float* out = (float*)d_out;

  dim3 grid(S_DIM * BLOCKS_PER_STACK);  // 512 blocks
  dim3 block(256);                      // 8 wave32s
  hipLaunchKernelGGL(grouped_mlp_kernel, grid, block, 0, stream,
                     x, W1, b1, W2, b2, lam, out);
}
